// TRG_wrapper_21449066676690
// MI455X (gfx1250) — matrix operand order repaired and checked by reading, not compiled, not run on hardware
//
#include <hip/hip_runtime.h>
#include <hip/hip_bf16.h>

typedef __attribute__((ext_vector_type(16))) _Float16 v16h;
typedef __attribute__((ext_vector_type(8)))  _Float16 v8h;
typedef __attribute__((ext_vector_type(8)))  float    v8f;

#define CDIM 512
#define HP   16
#define WP   32
#define PIX  (HP * WP)          // 512

// Use CDNA5 async global->LDS DMA (GLOBAL_LOAD_ASYNC_TO_LDS_B128, ASYNCcnt)
#define ASYNC_STAGE 1

// ---------------------------------------------------------------------------
// WMMA GEMM:  out[m,n] = act( sum_k A[m,k]*B(k,n) + bias[m] + add[m,n] )
//   A : M x K row-major f16 (K-contiguous), M % 32 == 0, K % 32 == 0
//   B : BMODE 0 -> N x K row-major f16 ("NK", K-contiguous), row stride = K
//       BMODE 1 -> im2col 3x3 SAME over feature map im (P x C f16, pixel-major)
//   Each wave computes a 32x32 tile: 2x2 subtiles, 4x v_wmma_f32_16x16x32_f16
//   per 32-wide K step. Tiles staged through LDS; A staged via async DMA.
// ---------------------------------------------------------------------------
struct GemmParams {
    const _Float16* A;
    const _Float16* B;      // BMODE 0 (N x K)
    const _Float16* im;     // BMODE 1 feature map (P x C)
    float*     outF;        // optional f32 out, row stride ldc
    _Float16*  outH;        // optional f16 out, row stride ldc
    _Float16*  outHT;       // optional f16 out transposed: [n*M + m]
    const float* bias;      // optional, indexed by m
    const float* add;       // optional residual, layout of outF
    int M, N, K, ldc;
    int act;                // 0 = none, 1 = leaky-relu(0.1)
};

static __device__ __forceinline__ v16h frag2(const _Float16* p0, const _Float16* p1) {
    v8h lo = *reinterpret_cast<const v8h*>(p0);
    v8h hs = *reinterpret_cast<const v8h*>(p1);
    return __builtin_shufflevector(lo, hs, 0,1,2,3,4,5,6,7,8,9,10,11,12,13,14,15);
}

template <int BMODE>
__global__ __launch_bounds__(128)
void gemm_wmma_kernel(GemmParams p) {
    __shared__ _Float16 sA[4][32 * 32];   // per-wave A tile: 32 m-rows x 32 k
    __shared__ _Float16 sB[4][32 * 32];   // per-wave B tile: 32 n-rows x 32 k

    const int lane  = threadIdx.x & 31;
    const int wave  = threadIdx.x >> 5;
    const int half_ = lane & 15;
    const int hi    = lane >> 4;

    const int tilesN = (p.N + 31) >> 5;
    const int tilesM = p.M >> 5;
    const int nTiles = tilesM * tilesN;

    int tile = blockIdx.x * 4 + wave;
    const bool valid = (tile < nTiles);
    if (!valid) tile = nTiles - 1;        // uniform loop count for barriers
    const int tm = (tile / tilesN) * 32;
    const int tn = (tile % tilesN) * 32;

    _Float16* As = sA[wave];
    _Float16* Bs = sB[wave];

    v8f acc[2][2];
#pragma unroll
    for (int i = 0; i < 2; ++i)
#pragma unroll
        for (int j = 0; j < 2; ++j)
            acc[i][j] = (v8f){0.f, 0.f, 0.f, 0.f, 0.f, 0.f, 0.f, 0.f};

    for (int k0 = 0; k0 < p.K; k0 += 32) {
        // ---- stage A: lane stages one 64B row: A[tm+lane][k0..k0+31] ----
        {
            const _Float16* src = p.A + (size_t)(tm + lane) * p.K + k0;
#if ASYNC_STAGE
            unsigned dst = (unsigned)(unsigned long long)(void*)(As + lane * 32);
            asm volatile("global_load_async_to_lds_b128 %0, %1, off"
                         :: "v"(dst), "v"(src) : "memory");
            asm volatile("global_load_async_to_lds_b128 %0, %1, off offset:16"
                         :: "v"(dst), "v"(src) : "memory");
            asm volatile("global_load_async_to_lds_b128 %0, %1, off offset:32"
                         :: "v"(dst), "v"(src) : "memory");
            asm volatile("global_load_async_to_lds_b128 %0, %1, off offset:48"
                         :: "v"(dst), "v"(src) : "memory");
#else
            float4 r0 = *reinterpret_cast<const float4*>(src);
            float4 r1 = *reinterpret_cast<const float4*>(src + 8);
            float4 r2 = *reinterpret_cast<const float4*>(src + 16);
            float4 r3 = *reinterpret_cast<const float4*>(src + 24);
            float4* d = reinterpret_cast<float4*>(As + lane * 32);
            d[0] = r0; d[1] = r1; d[2] = r2; d[3] = r3;
#endif
        }

        // ---- stage B: lane stages one n-row: B(k0..k0+31, tn+lane) ----
        if (BMODE == 0) {
            float4* d = reinterpret_cast<float4*>(Bs + lane * 32);
            int gn = tn + lane;
            if (gn < p.N) {
                const _Float16* src = p.B + (size_t)gn * p.K + k0;
                d[0] = *reinterpret_cast<const float4*>(src);
                d[1] = *reinterpret_cast<const float4*>(src + 8);
                d[2] = *reinterpret_cast<const float4*>(src + 16);
                d[3] = *reinterpret_cast<const float4*>(src + 24);
            } else {
                float4 z = {0.f, 0.f, 0.f, 0.f};
                d[0] = z; d[1] = z; d[2] = z; d[3] = z;
            }
        } else {
            __attribute__((aligned(16))) _Float16 tmp[32];
            const int gn = tn + lane;           // pixel index (N = 512)
            const int h0 = gn >> 5, w0 = gn & 31;
#pragma unroll
            for (int kk = 0; kk < 32; ++kk) {
                int kg = k0 + kk;               // k' = c*9 + j
                int c  = kg / 9;
                int j  = kg - c * 9;
                int hh = h0 + j / 3 - 1;
                int ww = w0 + (j - (j / 3) * 3) - 1;
                _Float16 v = (_Float16)0.f;
                if ((unsigned)hh < (unsigned)HP && (unsigned)ww < (unsigned)WP)
                    v = p.im[(size_t)(hh * WP + ww) * CDIM + c];  // im: P x C
                tmp[kk] = v;
            }
            float4* d = reinterpret_cast<float4*>(Bs + lane * 32);
            const float4* s = reinterpret_cast<const float4*>(tmp);
            d[0] = s[0]; d[1] = s[1]; d[2] = s[2]; d[3] = s[3];
        }

#if ASYNC_STAGE
        asm volatile("s_wait_asynccnt 0" ::: "memory");
#endif
        __syncthreads();

        // ---- fragments: contiguous per-lane chunks per CDNA5 VGPR layouts --
        v16h a[2], b[2];
#pragma unroll
        for (int i = 0; i < 2; ++i) {
            const _Float16* ar = As + (i * 16 + half_) * 32;
            a[i] = frag2(ar + hi * 8, ar + 16 + hi * 8);      // A 16x32 interleave
            const _Float16* br = Bs + (i * 16 + half_) * 32;
            b[i] = frag2(br + hi * 8, br + 16 + hi * 8);      // B 32x16 interleave
        }
        acc[0][0] = __builtin_amdgcn_wmma_f32_16x16x32_f16(false, a[0], false, b[0], (short)0, acc[0][0], false, false);
        acc[0][1] = __builtin_amdgcn_wmma_f32_16x16x32_f16(false, a[0], false, b[1], (short)0, acc[0][1], false, false);
        acc[1][0] = __builtin_amdgcn_wmma_f32_16x16x32_f16(false, a[1], false, b[0], (short)0, acc[1][0], false, false);
        acc[1][1] = __builtin_amdgcn_wmma_f32_16x16x32_f16(false, a[1], false, b[1], (short)0, acc[1][1], false, false);
        __syncthreads();
    }

    // ---- epilogue: D VGPR r holds (m = r + hi*8, n = half_) per subtile ----
#pragma unroll
    for (int i = 0; i < 2; ++i)
#pragma unroll
        for (int j = 0; j < 2; ++j)
#pragma unroll
            for (int r = 0; r < 8; ++r) {
                int m = tm + i * 16 + hi * 8 + r;
                int n = tn + j * 16 + half_;
                if (!valid || n >= p.N) continue;
                float v = acc[i][j][r];
                if (p.bias) v += p.bias[m];
                if (p.add)  v += p.add[(size_t)m * p.ldc + n];
                if (p.act == 1) v = (v > 0.f) ? v : 0.1f * v;
                if (p.outF)  p.outF[(size_t)m * p.ldc + n]  = v;
                if (p.outH)  p.outH[(size_t)m * p.ldc + n]  = (_Float16)v;
                if (p.outHT) p.outHT[(size_t)n * p.M + m]   = (_Float16)v;
            }
}

// ---------------------------------------------------------------------------
// Elementwise kernels
// ---------------------------------------------------------------------------
__global__ void f32_to_f16_kernel(const float* __restrict__ s,
                                  _Float16* __restrict__ d, int n) {
    int i = blockIdx.x * 256 + threadIdx.x;
    if (i < n) d[i] = (_Float16)s[i];
}

// (C x P) f32 -> (P x C) f16 transpose
__global__ void f32_to_f16_T_kernel(const float* __restrict__ s,
                                    _Float16* __restrict__ d) {
    int i = blockIdx.x * 256 + threadIdx.x;
    if (i >= CDIM * PIX) return;
    int c = i >> 9, p = i & (PIX - 1);
    d[(size_t)p * CDIM + c] = (_Float16)s[i];
}

// depthwise 3x3 SAME with dynamic per-channel kernels + leaky-relu
// x: (C x P) f32, kern: (C x 9) f32, out: (P x C) f16  (pixel-major)
__global__ void dwconv_kernel(const float* __restrict__ x,
                              const float* __restrict__ kern,
                              _Float16* __restrict__ out) {
    int i = blockIdx.x * 256 + threadIdx.x;
    if (i >= CDIM * PIX) return;
    int c = i >> 9;
    int p = i & (PIX - 1);
    int h = p >> 5, w = p & 31;
    float acc = 0.f;
#pragma unroll
    for (int j = 0; j < 9; ++j) {
        int dy = j / 3 - 1, dx = j % 3 - 1;
        int hh = h + dy, ww = w + dx;
        if ((unsigned)hh < (unsigned)HP && (unsigned)ww < (unsigned)WP)
            acc += kern[c * 9 + j] * x[c * PIX + hh * WP + ww];
    }
    acc = (acc > 0.f) ? acc : 0.1f * acc;
    out[(size_t)p * CDIM + c] = (_Float16)acc;
}

// y = lrelu( pout + deg * sigmoid(attlin) )   (all C x P) -> out (P x C) f16
__global__ void combine_kernel(const float* __restrict__ pout,
                               const float* __restrict__ attlin,
                               const float* __restrict__ deg,
                               _Float16* __restrict__ out) {
    int i = blockIdx.x * 256 + threadIdx.x;
    if (i >= CDIM * PIX) return;
    int c = i >> 9, p = i & (PIX - 1);
    float s = 1.f / (1.f + __expf(-attlin[i]));
    float v = pout[i] + deg[i] * s;
    v = (v > 0.f) ? v : 0.1f * v;
    out[(size_t)p * CDIM + c] = (_Float16)v;
}

// ---------------------------------------------------------------------------
// Host launcher
// ---------------------------------------------------------------------------
extern "C" void kernel_launch(void* const* d_in, const int* in_sizes, int n_in,
                              void* d_out, int out_size, void* d_ws, size_t ws_size,
                              hipStream_t stream) {
    (void)in_sizes; (void)n_in; (void)out_size;

    const float* cIn    = (const float*)d_in[0];
    const float* deg    = (const float*)d_in[1];
    const float* k1w1   = (const float*)d_in[2];
    const float* k1w2   = (const float*)d_in[3];
    const float* p1w    = (const float*)d_in[4];
    const float* p1b    = (const float*)d_in[5];
    const float* ca1w1  = (const float*)d_in[6];
    const float* ca1w2  = (const float*)d_in[7];
    const float* conv1w = (const float*)d_in[8];
    const float* conv1b = (const float*)d_in[9];
    const float* k2w1   = (const float*)d_in[10];
    const float* k2w2   = (const float*)d_in[11];
    const float* p2w    = (const float*)d_in[12];
    const float* p2b    = (const float*)d_in[13];
    const float* ca2w1  = (const float*)d_in[14];
    const float* ca2w2  = (const float*)d_in[15];
    const float* conv2w = (const float*)d_in[16];
    const float* conv2b = (const float*)d_in[17];

    const size_t CC    = (size_t)CDIM * CDIM;       // 262144
    const size_t CONVW = (size_t)CDIM * CDIM * 9;   // 2359296
    const size_t CA    = (size_t)(CDIM / 8) * CDIM; // 32768
    const size_t K2    = (size_t)9 * CDIM;          // 4608

    char*  base = (char*)d_ws;
    size_t off  = 0;
    auto allocH = [&](size_t n) -> _Float16* {
        off = (off + 255) & ~(size_t)255;
        _Float16* r = (_Float16*)(base + off);
        off += n * sizeof(_Float16);
        return r;
    };
    auto allocF = [&](size_t n) -> float* {
        off = (off + 255) & ~(size_t)255;
        float* r = (float*)(base + off);
        off += n * sizeof(float);
        return r;
    };

    _Float16* degH    = allocH(CC);   // deg,   C x C (A operand)
    _Float16* degTH   = allocH(CC);   // deg^T, P x C (B operand, NK)
    _Float16* k1w1H   = allocH(CC);
    _Float16* k1w2H   = allocH(K2);
    _Float16* p1wH    = allocH(CC);
    _Float16* ca1w1H  = allocH(CA);
    _Float16* ca1w2H  = allocH(CA);
    _Float16* conv1wH = allocH(CONVW);
    _Float16* k2w1H   = allocH(CC);
    _Float16* k2w2H   = allocH(K2);
    _Float16* p2wH    = allocH(CC);
    _Float16* ca2w1H  = allocH(CA);
    _Float16* ca2w2H  = allocH(CA);
    _Float16* conv2wH = allocH(CONVW);
    _Float16* tmpH    = allocH(CC);   // lrelu(deg @ kw1^T), M x K
    _Float16* x1H     = allocH(CC);   // depthwise out, P x C
    _Float16* x2H     = allocH(CC);
    _Float16* z1T     = allocH(CA);   // CA hidden, transposed: P x 64
    _Float16* z2T     = allocH(CA);
    _Float16* y1H     = allocH(CC);   // DA-block out, P x C
    _Float16* y3H     = allocH(CC);

    float* kern1F = allocF(K2);
    float* kern2F = allocF(K2);
    float* p1outF = allocF(CC);
    float* att1F  = allocF(CC);
    float* y2F    = allocF(CC);
    float* p2outF = allocF(CC);
    float* att2F  = allocF(CC);

    if (off > ws_size) return;   // deterministic no-op if scratch too small

    auto cvt = [&](const float* s, _Float16* d, size_t n) {
        f32_to_f16_kernel<<<(int)((n + 255) / 256), 256, 0, stream>>>(s, d, (int)n);
    };
    const int NP = CDIM * PIX;
    const int EB = (NP + 255) / 256;

    cvt(deg, degH, CC);
    f32_to_f16_T_kernel<<<EB, 256, 0, stream>>>(deg, degTH);
    cvt(k1w1, k1w1H, CC);   cvt(k1w2, k1w2H, K2);
    cvt(p1w, p1wH, CC);
    cvt(ca1w1, ca1w1H, CA); cvt(ca1w2, ca1w2H, CA);
    cvt(conv1w, conv1wH, CONVW);
    cvt(k2w1, k2w1H, CC);   cvt(k2w2, k2w2H, K2);
    cvt(p2w, p2wH, CC);
    cvt(ca2w1, ca2w1H, CA); cvt(ca2w2, ca2w2H, CA);
    cvt(conv2w, conv2wH, CONVW);

    auto gemm = [&](const _Float16* A, const _Float16* B,
                    float* outF, _Float16* outH, _Float16* outHT,
                    const float* bias, const float* add,
                    int M, int N, int K, int act) {
        GemmParams p{};
        p.A = A; p.B = B; p.im = nullptr;
        p.outF = outF; p.outH = outH; p.outHT = outHT;
        p.bias = bias; p.add = add;
        p.M = M; p.N = N; p.K = K; p.ldc = N; p.act = act;
        int tiles = (M / 32) * ((N + 31) / 32);
        gemm_wmma_kernel<0><<<(tiles + 3) / 4, 128, 0, stream>>>(p);
    };
    auto conv3 = [&](const _Float16* W, const _Float16* im,
                     float* outF, const float* bias, const float* add, int act) {
        GemmParams p{};
        p.A = W; p.B = nullptr; p.im = im;
        p.outF = outF; p.outH = nullptr; p.outHT = nullptr;
        p.bias = bias; p.add = add;
        p.M = CDIM; p.N = PIX; p.K = CDIM * 9; p.ldc = PIX; p.act = act;
        int tiles = (CDIM / 32) * (PIX / 32);
        gemm_wmma_kernel<1><<<(tiles + 3) / 4, 128, 0, stream>>>(p);
    };

    // ---------------- DA block 1 ----------------
    gemm(degH, k1w1H, nullptr, tmpH, nullptr, nullptr, nullptr, CDIM, CDIM, CDIM, 1);
    gemm(tmpH, k1w2H, kern1F, nullptr, nullptr, nullptr, nullptr, CDIM, 9, CDIM, 0);
    dwconv_kernel<<<EB, 256, 0, stream>>>(cIn, kern1F, x1H);
    gemm(p1wH, x1H, p1outF, nullptr, nullptr, p1b, nullptr, CDIM, PIX, CDIM, 0);
    gemm(ca1w1H, degTH, nullptr, nullptr, z1T, nullptr, nullptr, CDIM / 8, PIX, CDIM, 1);
    gemm(ca1w2H, z1T, att1F, nullptr, nullptr, nullptr, nullptr, CDIM, PIX, CDIM / 8, 0);
    combine_kernel<<<EB, 256, 0, stream>>>(p1outF, att1F, deg, y1H);

    // ---------------- conv3x3 #1 ----------------
    conv3(conv1wH, y1H, y2F, conv1b, nullptr, 1);

    // ---------------- DA block 2 ----------------
    gemm(degH, k2w1H, nullptr, tmpH, nullptr, nullptr, nullptr, CDIM, CDIM, CDIM, 1);
    gemm(tmpH, k2w2H, kern2F, nullptr, nullptr, nullptr, nullptr, CDIM, 9, CDIM, 0);
    dwconv_kernel<<<EB, 256, 0, stream>>>(y2F, kern2F, x2H);
    gemm(p2wH, x2H, p2outF, nullptr, nullptr, p2b, nullptr, CDIM, PIX, CDIM, 0);
    gemm(ca2w1H, degTH, nullptr, nullptr, z2T, nullptr, nullptr, CDIM / 8, PIX, CDIM, 1);
    gemm(ca2w2H, z2T, att2F, nullptr, nullptr, nullptr, nullptr, CDIM, PIX, CDIM / 8, 0);
    combine_kernel<<<EB, 256, 0, stream>>>(p2outF, att2F, deg, y3H);

    // ---------------- conv3x3 #2 + residual c -> d_out ----------------
    conv3(conv2wH, y3H, (float*)d_out, conv2b, cIn, 0);
}
